// Quantizer_56023553409086
// MI455X (gfx1250) — matrix operand order repaired and checked by reading, not compiled, not run on hardware
//
#include <hip/hip_runtime.h>

// ---------------------------------------------------------------------------
// VQ quantizer for MI455X (gfx1250, wave32, WMMA).
// B=16, L=2048, D=256, K=1024.  GEMM via v_wmma_f32_16x16x32_bf16 with
// bf16 hi/lo split (3 WMMAs per K-chunk) for ~fp32-accurate argmin.
// ---------------------------------------------------------------------------

typedef __bf16 bf16_t;
typedef bf16_t v16bf __attribute__((ext_vector_type(16)));
typedef float  v8f   __attribute__((ext_vector_type(8)));
typedef unsigned short v16us __attribute__((ext_vector_type(16)));

#define D_DIM        256
#define K_CODES      1024
#define L_SEQ        2048
#define B_BATCH      16
#define ROWS_PER_WG  128           // 8 waves * 16 rows
#define CHUNK_CODES  64            // 4 tiles of 16 codes per LDS chunk
#define NUM_CHUNKS   (K_CODES / CHUNK_CODES)

// split fp32 -> bf16 hi (truncate) + bf16 lo (residual); hi+lo carries ~16
// mantissa bits so hi*hi + hi*lo + lo*hi reproduces fp32 dot products.
__device__ __forceinline__ void splitf(float f, unsigned& h, unsigned& l) {
  unsigned u = __float_as_uint(f);
  h = u >> 16;
  float fh = __uint_as_float(u & 0xffff0000u);
  l = __float_as_uint(f - fh) >> 16;
}

// -------------------- prep: cb_sq[k] = sum_d codebook[k,d]^2 ---------------
__global__ void vq_cbsq(const float* __restrict__ cb, float* __restrict__ cbsq) {
  const int k = blockIdx.x;
  const int t = threadIdx.x;
  float f = cb[(size_t)k * D_DIM + t];
  float s = f * f;
#pragma unroll
  for (int off = 16; off > 0; off >>= 1) s += __shfl_xor(s, off, 32);
  __shared__ float red[8];
  if ((t & 31) == 0) red[t >> 5] = s;
  __syncthreads();
  if (t == 0) {
    float tot = 0.f;
#pragma unroll
    for (int i = 0; i < 8; ++i) tot += red[i];
    cbsq[k] = tot;
  }
}

// -------------------- prep: denom[b] = sum_l mask[b,l] ---------------------
__global__ void vq_denom(const float* __restrict__ fmask, float* __restrict__ denom) {
  const int b = blockIdx.x;
  const int t = threadIdx.x;
  float s = 0.f;
  for (int i = t; i < L_SEQ; i += 256) s += fmask[(size_t)b * L_SEQ + i];
#pragma unroll
  for (int off = 16; off > 0; off >>= 1) s += __shfl_xor(s, off, 32);
  __shared__ float red[8];
  if ((t & 31) == 0) red[t >> 5] = s;
  __syncthreads();
  if (t == 0) {
    float tot = 0.f;
#pragma unroll
    for (int i = 0; i < 8; ++i) tot += red[i];
    denom[b] = tot;
  }
}

// -------------------- main: argmin + gather + loss partials ----------------
__global__ __launch_bounds__(256)
void vq_main(const float* __restrict__ features,
             const float* __restrict__ fmask,
             const float* __restrict__ codebook,
             const float* __restrict__ cbsq,
             float* __restrict__ outq,
             float* __restrict__ partials) {
  __shared__ uint4 lds4[4096];          // 64 KB: 1024 slots * 64B (32B hi + 32B lo)

  const int tid  = threadIdx.x;
  const int wave = tid >> 5;
  const int lane = tid & 31;
  const int g    = lane >> 4;           // half-wave group
  const int ln   = lane & 15;
  const int rowBase = blockIdx.x * ROWS_PER_WG + wave * 16;
  const int rowA    = rowBase + ln;

  // ---- preload A: 16 rows x 256 dims, bf16 hi/lo, contiguous-k A layout (k = 16g + j) ----
  // lanes 0-15 (g=0): K = {d0..d0+15} with d0 = kc*32
  // lanes 16-31 (g=1): same +16
  v16bf a_hi[8], a_lo[8];
#pragma unroll
  for (int kc = 0; kc < 8; ++kc) {
    const int d0 = kc * 32 + g * 16;
    const float* fr = features + (size_t)rowA * D_DIM;
    float4 f0 = *(const float4*)(fr + d0);
    float4 f1 = *(const float4*)(fr + d0 + 4);
    float4 f2 = *(const float4*)(fr + d0 + 8);
    float4 f3 = *(const float4*)(fr + d0 + 12);
    float fv[16] = {f0.x, f0.y, f0.z, f0.w, f1.x, f1.y, f1.z, f1.w,
                    f2.x, f2.y, f2.z, f2.w, f3.x, f3.y, f3.z, f3.w};
    v16us hu, lu;
#pragma unroll
    for (int j = 0; j < 16; ++j) {
      unsigned h, l;
      splitf(fv[j], h, l);
      hu[j] = (unsigned short)h;
      lu[j] = (unsigned short)l;
    }
    a_hi[kc] = __builtin_bit_cast(v16bf, hu);
    a_lo[kc] = __builtin_bit_cast(v16bf, lu);
  }

  float bestV[8];
  int   bestI[8];
#pragma unroll
  for (int r = 0; r < 8; ++r) { bestV[r] = 3.4e38f; bestI[r] = 0; }

  for (int ch = 0; ch < NUM_CHUNKS; ++ch) {
    const int chunkBase = ch * CHUNK_CODES;
    __syncthreads();
    // ---- stage 64 codes into LDS, bf16 hi/lo, contiguous-k B layout per slot ------
    // slot = (tile*8 + kc)*32 + lane; lane holds code (chunk + tile*16 + ln)
    // with 16 contiguous K values starting at kc*32 + g*16.
#pragma unroll
    for (int s = 0; s < 4; ++s) {
      const int slot   = tid + s * 256;          // 0..1023
      const int slLane = slot & 31;
      const int slKc   = (slot >> 5) & 7;
      const int slT    = slot >> 8;              // 0..3
      const int code   = chunkBase + slT * 16 + (slLane & 15);
      const int d0     = slKc * 32 + (slLane >> 4) * 16;
      const float* cr = codebook + (size_t)code * D_DIM + d0;
      float4 c0 = *(const float4*)(cr);
      float4 c1 = *(const float4*)(cr + 4);
      float4 c2 = *(const float4*)(cr + 8);
      float4 c3 = *(const float4*)(cr + 12);
      float cv[16] = {c0.x, c0.y, c0.z, c0.w, c1.x, c1.y, c1.z, c1.w,
                      c2.x, c2.y, c2.z, c2.w, c3.x, c3.y, c3.z, c3.w};
      unsigned hw[8], lw[8];
#pragma unroll
      for (int j = 0; j < 8; ++j) {
        unsigned h0, l0, h1, l1;
        splitf(cv[2 * j],     h0, l0);
        splitf(cv[2 * j + 1], h1, l1);
        hw[j] = h0 | (h1 << 16);
        lw[j] = l0 | (l1 << 16);
      }
      uint4* dst = &lds4[slot * 4];
      dst[0] = make_uint4(hw[0], hw[1], hw[2], hw[3]);
      dst[1] = make_uint4(hw[4], hw[5], hw[6], hw[7]);
      dst[2] = make_uint4(lw[0], lw[1], lw[2], lw[3]);
      dst[3] = make_uint4(lw[4], lw[5], lw[6], lw[7]);
    }
    __syncthreads();
    // ---- 4 tiles of 16 codes: 24 WMMAs each (hi*hi + hi*lo + lo*hi) ------
#pragma unroll
    for (int t = 0; t < 4; ++t) {
      const int codeN = chunkBase + t * 16 + ln;
      const float csq = cbsq[codeN];
      v8f c = {0.f, 0.f, 0.f, 0.f, 0.f, 0.f, 0.f, 0.f};
#pragma unroll
      for (int kc = 0; kc < 8; ++kc) {
        const v16bf* bp = (const v16bf*)&lds4[((t * 8 + kc) * 32 + lane) * 4];
        v16bf bh = bp[0];
        v16bf bl = bp[1];
        c = __builtin_amdgcn_wmma_f32_16x16x32_bf16(false, a_hi[kc], false, bh,
                                                    (short)0, c, false, false);
        c = __builtin_amdgcn_wmma_f32_16x16x32_bf16(false, a_hi[kc], false, bl,
                                                    (short)0, c, false, false);
        c = __builtin_amdgcn_wmma_f32_16x16x32_bf16(false, a_lo[kc], false, bh,
                                                    (short)0, c, false, false);
      }
      // dist2[n,k] (minus const ||x||^2) = ||w||^2 - 2*score
#pragma unroll
      for (int r = 0; r < 8; ++r) {
        float dist = csq - 2.0f * c[r];
        if (dist < bestV[r] || (dist == bestV[r] && codeN < bestI[r])) {
          bestV[r] = dist;
          bestI[r] = codeN;
        }
      }
    }
  }

  // ---- argmin reduce across the 16 lanes that share each row -------------
#pragma unroll
  for (int r = 0; r < 8; ++r) {
    float v = bestV[r];
    int   i = bestI[r];
#pragma unroll
    for (int off = 8; off > 0; off >>= 1) {
      float ov = __shfl_xor(v, off, 16);
      int   oi = __shfl_xor(i, off, 16);
      if (ov < v || (ov == v && oi < i)) { v = ov; i = oi; }
    }
    bestI[r] = i;   // lanes 0-15: row rowBase+r ; lanes 16-31: rowBase+8+r
  }

  // ---- gather codebook[idx] -> out, accumulate masked squared error ------
  float lossAcc = 0.f;
#pragma unroll
  for (int r = 0; r < 8; ++r) {
    const int idx0 = __shfl(bestI[r], 0, 32);
    const int idx1 = __shfl(bestI[r], 16, 32);
#pragma unroll
    for (int h = 0; h < 2; ++h) {
      const int row = rowBase + r + h * 8;
      const int idx = h ? idx1 : idx0;
      const float4* fp = (const float4*)(features + (size_t)row * D_DIM) + lane * 2;
      const float4* wp = (const float4*)(codebook + (size_t)idx * D_DIM) + lane * 2;
      float4* op = (float4*)(outq + (size_t)row * D_DIM) + lane * 2;
      float4 a0 = fp[0], a1 = fp[1];
      float4 w0 = wp[0], w1 = wp[1];
      op[0] = w0;
      op[1] = w1;
      float dx, sq = 0.f;
      dx = a0.x - w0.x; sq += dx * dx;
      dx = a0.y - w0.y; sq += dx * dx;
      dx = a0.z - w0.z; sq += dx * dx;
      dx = a0.w - w0.w; sq += dx * dx;
      dx = a1.x - w1.x; sq += dx * dx;
      dx = a1.y - w1.y; sq += dx * dx;
      dx = a1.z - w1.z; sq += dx * dx;
      dx = a1.w - w1.w; sq += dx * dx;
      lossAcc += sq * fmask[row];
    }
  }
#pragma unroll
  for (int off = 16; off > 0; off >>= 1) lossAcc += __shfl_xor(lossAcc, off, 32);
  __syncthreads();                       // LDS no longer needed for B tiles
  float* red = (float*)lds4;
  if (lane == 0) red[wave] = lossAcc;
  __syncthreads();
  if (tid == 0) {
    float tot = 0.f;
#pragma unroll
    for (int w = 0; w < 8; ++w) tot += red[w];
    partials[blockIdx.x] = tot;          // deterministic: no float atomics
  }
}

// -------------------- finalize: losses (quant == commit forward) -----------
__global__ void vq_final(const float* __restrict__ partials,
                         const float* __restrict__ denom,
                         float* __restrict__ outLoss) {
  const int b = threadIdx.x;
  if (b < B_BATCH) {
    float s = 0.f;
#pragma unroll
    for (int w = 0; w < 16; ++w) s += partials[b * 16 + w];  // 16 WGs per batch
    const float q = s / denom[b];
    outLoss[b] = q;                // quant_loss
    outLoss[B_BATCH + b] = q;      // commit_loss (identical forward value)
  }
}

extern "C" void kernel_launch(void* const* d_in, const int* in_sizes, int n_in,
                              void* d_out, int out_size, void* d_ws, size_t ws_size,
                              hipStream_t stream) {
  (void)in_sizes; (void)n_in; (void)out_size; (void)ws_size;
  const float* features = (const float*)d_in[0];   // [16,2048,256]
  const float* fmask    = (const float*)d_in[1];   // [16,2048]
  const float* codebook = (const float*)d_in[2];   // [1024,256]
  float* out = (float*)d_out;                      // quantized_st | quant | commit
  float* ws  = (float*)d_ws;
  float* cbsq     = ws;                 // 1024 floats
  float* denom    = ws + K_CODES;       // 16 floats
  float* partials = ws + K_CODES + 16;  // 256 floats

  vq_cbsq <<<K_CODES, 256, 0, stream>>>(codebook, cbsq);
  vq_denom<<<B_BATCH, 256, 0, stream>>>(fmask, denom);
  vq_main <<<(B_BATCH * L_SEQ) / ROWS_PER_WG, 256, 0, stream>>>(
      features, fmask, codebook, cbsq, out, partials);
  vq_final<<<1, 32, 0, stream>>>(partials, denom,
                                 out + (size_t)B_BATCH * L_SEQ * D_DIM);
}
